// SparseLinear_72696616452696
// MI455X (gfx1250) — matrix operand order repaired and checked by reading, not compiled, not run on hardware
//
#include <hip/hip_runtime.h>

typedef __attribute__((ext_vector_type(4)))  float  v4f;
typedef __attribute__((ext_vector_type(8)))  float  v8f;
typedef __attribute__((ext_vector_type(2)))  __bf16 v2bf;
typedef __attribute__((ext_vector_type(4)))  __bf16 v4bf;
typedef __attribute__((ext_vector_type(8)))  __bf16 v8bf;
typedef __attribute__((ext_vector_type(16))) __bf16 v16bf;

#define M_DIM 4096
#define K_DIM 8192
#define N_DIM 4096

#define TM 256
#define TN 128
#define TK 32
#define NSTEPS (K_DIM / TK)

// y = x @ (mask * w), fused mask, bf16 WMMA, double-buffered LDS pipeline.
// Block: 256 threads (8 wave32) -> 256x128 output tile.
// Wave (wavem = wave&3, waven = wave>>2) -> 64x64 sub-tile = 4x4 WMMA tiles.
__global__ __launch_bounds__(256, 1) void sparse_linear_wmma_kernel(
    const float* __restrict__ X,
    const float* __restrict__ MSK,
    const float* __restrict__ W,
    float* __restrict__ OUT)
{
    __shared__ __bf16 As[2][TM * TK];   // 16 KB / buffer, row-major [m][k]
    __shared__ __bf16 Bs[2][TN * TK];   //  8 KB / buffer, transposed [n][k]

    const int tid   = threadIdx.x;
    const int lane  = tid & 31;
    const int wave  = tid >> 5;
    const int wavem = wave & 3;          // 4 waves along M: 64 rows each
    const int waven = wave >> 2;         // 2 waves along N: 64 cols each
    const int half  = lane >> 4;         // lane half-group (0/1)
    const int lrow  = lane & 15;         // M (A) / N (B) index inside 16-tile
    const int akoff = half * 16;         // A: K chunk base (0 or 16)
    const int bkoff = half * 16;         // B: K chunk base (0 or 16)

    const int tile_m = blockIdx.y * TM;
    const int tile_n = blockIdx.x * TN;

    v8f acc[4][4] = {};                  // 16 accum tiles * 8 VGPRs = 128 VGPRs

    // global->register staging (64 VGPRs): issued one K-step ahead
    v4f xs[8];                           // 256x32 x tile: 8 float4 / thread
    v4f ws[4], ms[4];                    // 32x128 w/mask tile: k-pairs x 4n

    auto g2r = [&](int k0) {
        #pragma unroll
        for (int i = 0; i < 8; ++i) {
            int idx = tid + i * 256;               // 0..2047 float4 slots
            int r   = idx >> 3;                    // row 0..255
            int c4  = idx & 7;                     // float4 col 0..7
            xs[i] = *(const v4f*)(X + (size_t)(tile_m + r) * K_DIM + k0 + c4 * 4);
        }
        #pragma unroll
        for (int i = 0; i < 2; ++i) {
            int idx = tid + i * 256;               // 0..511 pair-slots
            int kp  = idx >> 5;                    // k-pair 0..15 -> k rows 2kp,2kp+1
            int n4  = idx & 31;                    // float4 col 0..31
            size_t g = (size_t)(k0 + 2 * kp) * N_DIM + tile_n + n4 * 4;
            ws[2 * i + 0] = *(const v4f*)(W + g);
            ws[2 * i + 1] = *(const v4f*)(W + g + N_DIM);
            ms[2 * i + 0] = *(const v4f*)(MSK + g);
            ms[2 * i + 1] = *(const v4f*)(MSK + g + N_DIM);
        }
    };

    auto r2l = [&](int buf) {
        __bf16* a = As[buf];
        __bf16* b = Bs[buf];
        #pragma unroll
        for (int i = 0; i < 8; ++i) {
            int idx = tid + i * 256;
            int r   = idx >> 3;
            int c4  = idx & 7;
            v4bf xb = { (__bf16)xs[i].x, (__bf16)xs[i].y,
                        (__bf16)xs[i].z, (__bf16)xs[i].w };
            *(v4bf*)(&a[r * TK + c4 * 4]) = xb;
        }
        #pragma unroll
        for (int i = 0; i < 2; ++i) {
            int idx = tid + i * 256;
            int kp  = idx >> 5;
            int n4  = idx & 31;
            v4f p0 = ws[2 * i + 0] * ms[2 * i + 0];  // fused mask, k row 2kp
            v4f p1 = ws[2 * i + 1] * ms[2 * i + 1];  // fused mask, k row 2kp+1
            #pragma unroll
            for (int e = 0; e < 4; ++e) {
                v2bf pk = { (__bf16)p0[e], (__bf16)p1[e] };   // pack (k, k+1)
                *(v2bf*)(&b[(n4 * 4 + e) * TK + 2 * kp]) = pk;
            }
        }
    };

    auto compute = [&](int buf) {
        const __bf16* a = As[buf];
        const __bf16* b = Bs[buf];
        v16bf afrag[4];
        #pragma unroll
        for (int i = 0; i < 4; ++i) {
            // A 16x32 bf16: half 0 -> K[0:16], half 1 -> K[16:32] (matches B)
            const __bf16* p = &a[(wavem * 64 + i * 16 + lrow) * TK + akoff];
            v8bf lo = *(const v8bf*)(p);
            v8bf hi = *(const v8bf*)(p + 8);
            afrag[i] = __builtin_shufflevector(lo, hi,
                0,1,2,3,4,5,6,7,8,9,10,11,12,13,14,15);
        }
        v16bf bfrag[4];
        #pragma unroll
        for (int j = 0; j < 4; ++j) {
            // B 32x16 bf16: half 0 -> K[0:16], half 1 -> K[16:32], col = lane&15
            bfrag[j] = *(const v16bf*)(&b[(waven * 64 + j * 16 + lrow) * TK + bkoff]);
        }
        #pragma unroll
        for (int i = 0; i < 4; ++i)
            #pragma unroll
            for (int j = 0; j < 4; ++j)
                acc[i][j] = __builtin_amdgcn_wmma_f32_16x16x32_bf16(
                    false, afrag[i], false, bfrag[j],
                    (short)0, acc[i][j], false, false);
    };

    // ---- software pipeline: 1 barrier per K-step, buffers alternate ----
    g2r(0);
    r2l(0);
    __syncthreads();

    for (int s = 0; s < NSTEPS; ++s) {
        const int cur = s & 1;
        if (s + 1 < NSTEPS) g2r((s + 1) * TK);   // issue next-tile global loads
        compute(cur);                            // ds_load + 16x v_wmma
        if (s + 1 < NSTEPS) r2l(cur ^ 1);        // convert + store into other buffer
        __syncthreads();
    }

    // ---- writeback, non-temporal (C has zero reuse; keep L2 for w/mask stream) ----
    #pragma unroll
    for (int i = 0; i < 4; ++i)
        #pragma unroll
        for (int j = 0; j < 4; ++j)
            #pragma unroll
            for (int r = 0; r < 8; ++r) {
                int row = tile_m + wavem * 64 + i * 16 + half * 8 + r;
                int col = tile_n + waven * 64 + j * 16 + lrow;
                __builtin_nontemporal_store(acc[i][j][r],
                                            OUT + (size_t)row * N_DIM + col);
            }
}

extern "C" void kernel_launch(void* const* d_in, const int* in_sizes, int n_in,
                              void* d_out, int out_size, void* d_ws, size_t ws_size,
                              hipStream_t stream) {
    const float* x    = (const float*)d_in[0];   // (4096, 8192)
    const float* mask = (const float*)d_in[1];   // (8192, 4096)
    const float* w    = (const float*)d_in[2];   // (8192, 4096)
    float* out        = (float*)d_out;           // (4096, 4096)

    dim3 grid(N_DIM / TN, M_DIM / TM);           // 32 x 16 = 512 blocks
    dim3 block(256);
    sparse_linear_wmma_kernel<<<grid, block, 0, stream>>>(x, mask, w, out);
}
